// BoundaryLoss3D_14860586844746
// MI455X (gfx1250) — compile-verified, matrix operand order checked by reading, not run on hardware
//
#include <hip/hip_runtime.h>
#include <hip/hip_bf16.h>

#define BIGF 1.0e12f

namespace {
constexpr int kN = 2, kC = 4, kD = 64, kH = 128, kW = 128;
constexpr int kHW  = kH * kW;          // 16384
constexpr int kDHW = kD * kH * kW;     // 1048576
constexpr int kLinesW   = kN * kD * kH;        // 16384 W-lines per class
constexpr int kBlocksD  = kN * kH * (kW / 32); // 1024
constexpr int kBlocksH  = kN * kD * (kW / 32); // 512
constexpr int kTotalBlockSums = kC * kLinesW;  // 65536
}

typedef __attribute__((ext_vector_type(2)))  float        v2f;
typedef __attribute__((ext_vector_type(8)))  float        v8f;
typedef __attribute__((ext_vector_type(16))) _Float16     v16h;
typedef __attribute__((ext_vector_type(4)))  unsigned int v4u;
typedef __attribute__((ext_vector_type(8)))  int          v8i;
typedef __attribute__((ext_vector_type(4)))  int          v4i;

// ---------------------------------------------------------------------------
// Tensor Data Mover staging: one 2D descriptor pulls a [rows x 32-elem] tile
// (rows strided by row_stride elements) from global into contiguous LDS,
// tracked on TENSORcnt. Guarded so any toolchain still compiles (fallback =
// per-thread VMEM loads + ds stores).
// ---------------------------------------------------------------------------
#if __has_builtin(__builtin_amdgcn_tensor_load_to_lds) && \
    __has_builtin(__builtin_amdgcn_s_wait_tensorcnt)
#define BL3D_HAVE_TDM 1
#if defined(__has_include)
#if __has_include(<hip/amd_detail/amd_gfx1250_TDM.h>)
#define BL3D_TDM_6ARG 1
#endif
#endif
#endif

#ifdef BL3D_HAVE_TDM
__device__ __forceinline__ void tdm_load_tile_2d(const float* gptr, void* lds_ptr,
                                                 unsigned tile_w, unsigned tile_h,
                                                 unsigned row_stride_elems) {
  const unsigned long long ga = (unsigned long long)(uintptr_t)gptr;
  const unsigned la = (unsigned)(uintptr_t)lds_ptr;  // low 32 bits = LDS offset
  v4u g0;
  g0[0] = 1u;                                        // count=1, is_restore=0
  g0[1] = la;                                        // lds_addr
  g0[2] = (unsigned)(ga & 0xFFFFFFFFu);              // global_addr[31:0]
  g0[3] = (unsigned)((ga >> 32) & 0x01FFFFFFu)       // global_addr[56:32]
        | (2u << 30);                                // type = 2 ("image")
  v8i g1;
  g1[0] = (int)(2u << 16);                           // data_size=2 (4B), mask=0
  g1[1] = (int)((tile_w & 0xFFFFu) << 16);           // tensor_dim0[15:0]
  g1[2] = (int)((tile_w >> 16) | ((tile_h & 0xFFFFu) << 16));  // dim0[31:16]|dim1[15:0]
  g1[3] = (int)((tile_h >> 16) | (tile_w << 16));    // dim1[31:16] | tile_dim0
  g1[4] = (int)(tile_h & 0xFFFFu);                   // tile_dim1; tile_dim2=0
  g1[5] = (int)row_stride_elems;                     // tensor_dim0_stride[31:0]
  g1[6] = 0;                                         // stride[47:32], dim1_stride lo
  g1[7] = 0;
  v4i gz = {};                                       // groups 2/3 unused (<=2D)
#ifdef BL3D_TDM_6ARG
  v8i gz8 = {};
  __builtin_amdgcn_tensor_load_to_lds(g0, g1, gz, gz, gz8, 0);
#else
  __builtin_amdgcn_tensor_load_to_lds(g0, g1, gz, gz, 0);
#endif
}
#endif  // BL3D_HAVE_TDM

// ---------------------------------------------------------------------------
// Wave-level sum via the matrix engine (ones-matmul trick), f32 16x16x4 WMMA.
// D = A x ones + 0 => D[m,n] = x_m + x_{m+16}; summing the 8 accumulator
// VGPRs gives half-wave sums in lanes {0..15} and {16..31}.
// ---------------------------------------------------------------------------
__device__ __forceinline__ float wave_halfsum_wmma(float x) {
#if __has_builtin(__builtin_amdgcn_wmma_f32_16x16x4_f32)
  v2f a; a[0] = x;    a[1] = 0.0f;
  v2f b; b[0] = 1.0f; b[1] = 1.0f;
  v8f acc = {};
  acc = __builtin_amdgcn_wmma_f32_16x16x4_f32(false, a, false, b, (short)0, acc,
                                              false, false);
#else
  v16h a = {};
  a[0] = (_Float16)x;
  v16h b;
#pragma unroll
  for (int k = 0; k < 16; ++k) b[k] = (_Float16)1.0f;
  v8f acc = {};
  acc = __builtin_amdgcn_wmma_f32_16x16x32_f16(false, a, false, b, (short)0, acc,
                                               false, false);
#endif
  return acc[0] + acc[1] + acc[2] + acc[3] + acc[4] + acc[5] + acc[6] + acc[7];
}

// Pass 0: f = (targets==c ? 0 : BIG) fused with the N-axis (length-2) pass:
// out[n] = min(f[n], f[1-n] + 1).
__global__ void bl3d_init_npass(const int* __restrict__ targets,
                                float* __restrict__ f, int c) {
  int idx = blockIdx.x * blockDim.x + threadIdx.x;  // grid == kDHW exactly
  int t0 = targets[idx];
  int t1 = targets[kDHW + idx];
  float f0 = (t0 == c) ? 0.0f : BIGF;
  float f1 = (t1 == c) ? 0.0f : BIGF;
  f[idx]        = fminf(f0, f1 + 1.0f);
  f[kDHW + idx] = fminf(f1, f0 + 1.0f);
}

// D-axis pass (L=64, element stride H*W). Tile [64 d x 32 w] staged into LDS
// by the TDM (rows are 128B segments strided 64KB apart -> ideal DMA shape).
__global__ void bl3d_edt_d(float* __restrict__ f) {
  __shared__ float tile[kD * 32];
  const int b   = blockIdx.x;        // (n, h, w-tile)
  const int n   = b >> 9;            // / (kH*4)
  const int rem = b & 511;
  const int h   = rem >> 2;
  const int w0  = (rem & 3) << 5;
  const int tid = threadIdx.x;       // 256
  const int w   = tid & 31;
  const int q   = tid >> 5;          // 0..7
  const size_t tileBase = (size_t)n * kDHW + (size_t)h * kW + w0;
  const size_t base     = tileBase + w;
#ifdef BL3D_HAVE_TDM
  if (tid < 32) {                    // wave 0 (uniform branch) issues the DMA
    tdm_load_tile_2d(&f[tileBase], tile, 32u, (unsigned)kD, (unsigned)kHW);
    __builtin_amdgcn_s_wait_tensorcnt(0);
  }
#else
#pragma unroll
  for (int i = 0; i < kD / 8; ++i) {
    int d = q + (i << 3);
    tile[d * 32 + w] = f[base + (size_t)d * kHW];
  }
#endif
  __syncthreads();
  float m[kD / 8];
#pragma unroll
  for (int i = 0; i < kD / 8; ++i) m[i] = 3.4e38f;
  for (int j = 0; j < kD; ++j) {
    float v = tile[j * 32 + w];
#pragma unroll
    for (int i = 0; i < kD / 8; ++i) {
      float dj = (float)(q + (i << 3) - j);
      m[i] = fminf(m[i], fmaf(dj, dj, v));
    }
  }
#pragma unroll
  for (int i = 0; i < kD / 8; ++i) {
    int d = q + (i << 3);
    f[base + (size_t)d * kHW] = m[i];
  }
}

// H-axis pass (L=128, element stride W). Tile [128 h x 32 w] in LDS (16 KB),
// staged by the TDM (rows are 128B segments strided 512B apart).
__global__ void bl3d_edt_h(float* __restrict__ f) {
  __shared__ float tile[kH * 32];
  const int b   = blockIdx.x;        // (n, d, w-tile)
  const int n   = b >> 8;            // / (kD*4)
  const int rem = b & 255;
  const int d   = rem >> 2;
  const int w0  = (rem & 3) << 5;
  const int tid = threadIdx.x;       // 256
  const int w   = tid & 31;
  const int q   = tid >> 5;          // 0..7
  const size_t tileBase = (size_t)n * kDHW + (size_t)d * kHW + w0;
  const size_t base     = tileBase + w;
#ifdef BL3D_HAVE_TDM
  if (tid < 32) {
    tdm_load_tile_2d(&f[tileBase], tile, 32u, (unsigned)kH, (unsigned)kW);
    __builtin_amdgcn_s_wait_tensorcnt(0);
  }
#else
#pragma unroll
  for (int i = 0; i < kH / 8; ++i) {
    int h = q + (i << 3);
    tile[h * 32 + w] = f[base + (size_t)h * kW];
  }
#endif
  __syncthreads();
  float m[kH / 8];
#pragma unroll
  for (int i = 0; i < kH / 8; ++i) m[i] = 3.4e38f;
  for (int j = 0; j < kH; ++j) {
    float v = tile[j * 32 + w];
#pragma unroll
    for (int i = 0; i < kH / 8; ++i) {
      float dj = (float)(q + (i << 3) - j);
      m[i] = fminf(m[i], fmaf(dj, dj, v));
    }
  }
#pragma unroll
  for (int i = 0; i < kH / 8; ++i) {
    int h = q + (i << 3);
    f[base + (size_t)h * kW] = m[i];
  }
}

// W-axis pass fused with sqrt, onehot-masked dot with logits, and a
// WMMA-based block reduction. One 128-thread block (4 waves) per line.
__global__ void bl3d_edt_w_reduce(const float* __restrict__ f,
                                  const float* __restrict__ logits,
                                  const int* __restrict__ targets,
                                  float* __restrict__ blocksums, int c) {
  __shared__ float line[kW];
  __shared__ float partial[8];
  const int b   = blockIdx.x;        // 16384 lines
  const int n   = b >> 13;
  const int rem = b & 8191;
  const int d   = rem >> 7;
  const int h   = rem & 127;
  const int i   = threadIdx.x;       // 0..127
  const size_t lineBase = (size_t)n * kDHW + (size_t)d * kHW + (size_t)h * kW;
  const size_t lIdx = (size_t)(((n * kC + c) * kD + d) * kH + h) * kW + i;
  __builtin_prefetch(&logits[lIdx], 0, 0);  // global_prefetch_b8
  line[i] = f[lineBase + i];                // contiguous 512B, already optimal
  __syncthreads();
  float m = 3.4e38f;
  for (int j = 0; j < kW; ++j) {
    float dj = (float)(i - j);
    m = fminf(m, fmaf(dj, dj, line[j]));    // broadcast LDS read, conflict-free
  }
  float dist = sqrtf(m);
  float contrib = (targets[lineBase + i] == c) ? logits[lIdx] * dist : 0.0f;

  // non-divergent wave reduction on the matrix engine (EXEC all ones)
  float s = wave_halfsum_wmma(contrib);
  int lane = i & 31, wid = i >> 5;
  if (lane == 0)  partial[2 * wid]     = s;
  if (lane == 16) partial[2 * wid + 1] = s;
  __syncthreads();
  if (i == 0) {
    float acc = 0.0f;
#pragma unroll
    for (int k = 0; k < 8; ++k) acc += partial[k];
    blocksums[b] = acc;
  }
}

// Deterministic final reduction + mean scaling (1/(N*C*D*H*W) is a pow2).
__global__ void bl3d_final(const float* __restrict__ blocksums,
                           float* __restrict__ out) {
  __shared__ float sh[256];
  float acc = 0.0f;
  for (int idx = threadIdx.x; idx < kTotalBlockSums; idx += 256)
    acc += blocksums[idx];
  sh[threadIdx.x] = acc;
  __syncthreads();
  for (int s = 128; s > 0; s >>= 1) {
    if ((int)threadIdx.x < s) sh[threadIdx.x] += sh[threadIdx.x + s];
    __syncthreads();
  }
  if (threadIdx.x == 0)
    out[0] = sh[0] * (1.0f / (float)((size_t)kN * kC * kDHW));
}

extern "C" void kernel_launch(void* const* d_in, const int* in_sizes, int n_in,
                              void* d_out, int out_size, void* d_ws, size_t ws_size,
                              hipStream_t stream) {
  (void)in_sizes; (void)n_in; (void)out_size; (void)ws_size;
  const float* logits  = (const float*)d_in[0];
  const int*   targets = (const int*)d_in[1];
  float* f         = (float*)d_ws;                 // 8 MB, reused per class
  float* blocksums = f + (size_t)kN * kDHW;        // 65536 floats

  for (int c = 0; c < kC; ++c) {
    bl3d_init_npass<<<kDHW / 256, 256, 0, stream>>>(targets, f, c);
    bl3d_edt_d<<<kBlocksD, 256, 0, stream>>>(f);
    bl3d_edt_h<<<kBlocksH, 256, 0, stream>>>(f);
    bl3d_edt_w_reduce<<<kLinesW, kW, 0, stream>>>(f, logits, targets,
                                                  blocksums + c * kLinesW, c);
  }
  bl3d_final<<<1, 256, 0, stream>>>(blocksums, (float*)d_out);
}
